// SinoVietnameseTranslator_68169720922280
// MI455X (gfx1250) — matrix operand order repaired and checked by reading, not compiled, not run on hardware
//
#include <hip/hip_runtime.h>
#include <hip/hip_bf16.h>
#include <math.h>

// MI455X / gfx1250, wave32. Matrix ops via v_wmma_f32_16x16x32_f16.

typedef __attribute__((ext_vector_type(16))) _Float16 v16h;
typedef __attribute__((ext_vector_type(8)))  _Float16 v8h;
typedef __attribute__((ext_vector_type(4)))  _Float16 v4h;
typedef __attribute__((ext_vector_type(8)))  float    v8f;
typedef __attribute__((ext_vector_type(4)))  float    v4f;

#define TOKENS 1024   // B*S
#define DMODEL 512    // N of every trunk GEMM (compile-time!)
#define HHID   256
#define KSP    7

// ---------------------------------------------------------------------------
// Trunk GEMM: out[M,512] = A[M,K] @ W[K,512] (+bias, optional relu/residual)
// Tile: 128(M) x 64(N) per 256-thread WG (8 waves, 4x2), each wave 2x2 WMMA
// tiles, BK=32 staged via LDS as f16 (packed b64 stores).
// EPI: 0 = bias, 1 = bias+relu, 2 = bias+residual
// ---------------------------------------------------------------------------
template <int EPI>
__global__ __launch_bounds__(256)
void gemm_wmma(const float* __restrict__ A, const float* __restrict__ W,
               const float* __restrict__ bias, const float* __restrict__ res,
               float* __restrict__ out, int K)
{
    constexpr int N = DMODEL;                         // 512, compile-time
    __shared__ __align__(16) _Float16 As[128 * 40];   // [row][k], stride 40 halves
    __shared__ __align__(16) _Float16 Bs[64 * 40];    // [n][k] transposed

    const int tid  = threadIdx.x;
    const int m0   = blockIdx.y * 128;
    const int n0   = blockIdx.x * 64;
    const int wave = tid >> 5;
    const int lane = tid & 31;
    const int wm   = wave & 3;        // M offset within WG tile
    const int wn   = wave >> 2;       // N offset
    const int g    = lane >> 4;       // lane group
    const int l16  = lane & 15;

    v8f c[2][2];
    for (int mt = 0; mt < 2; ++mt)
        for (int nt = 0; nt < 2; ++nt)
            c[mt][nt] = (v8f){0.f, 0.f, 0.f, 0.f, 0.f, 0.f, 0.f, 0.f};

    for (int k0 = 0; k0 < K; k0 += 32) {
        // ---- stage A: 128x32 f32 -> f16 LDS, one b64 store per float4 ----
        {
            const int quad  = tid & 7;
            const int rbase = tid >> 3;
            #pragma unroll
            for (int r = 0; r < 4; ++r) {
                const int row = rbase + r * 32;
                v4f a4 = *(const v4f*)(A + (size_t)(m0 + row) * K + k0 + quad * 4);
                v4h pk;
                pk[0] = (_Float16)a4.x; pk[1] = (_Float16)a4.y;
                pk[2] = (_Float16)a4.z; pk[3] = (_Float16)a4.w;
                *(v4h*)(&As[row * 40 + quad * 4]) = pk;   // 8B aligned
            }
        }
        // ---- stage B transposed: pack 4 consecutive k per b64 store ----
        {
            const int n  = tid & 63;
            const int kq = (tid >> 6) * 4;             // 0,4,8,12
            #pragma unroll
            for (int r = 0; r < 2; ++r) {
                const int k = kq + r * 16;
                const float* wp = W + (size_t)(k0 + k) * N + n0 + n;
                v4h pk;
                pk[0] = (_Float16)wp[0 * N];
                pk[1] = (_Float16)wp[1 * N];
                pk[2] = (_Float16)wp[2 * N];
                pk[3] = (_Float16)wp[3 * N];
                *(v4h*)(&Bs[n * 40 + k]) = pk;         // 8B aligned (k mult of 4)
            }
        }
        __syncthreads();

        // ---- fragment loads (ISA 7.12.2 layouts) + WMMA ----
        v16h a[2], b[2];
        #pragma unroll
        for (int mt = 0; mt < 2; ++mt) {
            const int row = wm * 32 + mt * 16 + l16;
            const _Float16* p = &As[row * 40 + g * 8];
            ((v8h*)&a[mt])[0] = *(const v8h*)(p);        // K = g*8 .. g*8+7
            ((v8h*)&a[mt])[1] = *(const v8h*)(p + 16);   // K = 16+g*8 ..
        }
        #pragma unroll
        for (int nt = 0; nt < 2; ++nt) {
            const int col = wn * 32 + nt * 16 + l16;
            const _Float16* p = &Bs[col * 40 + g * 8];   // K = g*8 .. g*8+7, 16+g*8 ..
            ((v8h*)&b[nt])[0] = *(const v8h*)(p);
            ((v8h*)&b[nt])[1] = *(const v8h*)(p + 16);
        }
        #pragma unroll
        for (int mt = 0; mt < 2; ++mt)
            #pragma unroll
            for (int nt = 0; nt < 2; ++nt)
                c[mt][nt] = __builtin_amdgcn_wmma_f32_16x16x32_f16(
                    false, a[mt], false, b[nt], (short)0, c[mt][nt], false, false);
        __syncthreads();
    }

    // ---- epilogue (N compile-time -> shift addressing) ----
    #pragma unroll
    for (int mt = 0; mt < 2; ++mt) {
        #pragma unroll
        for (int nt = 0; nt < 2; ++nt) {
            const int n  = n0 + wn * 32 + nt * 16 + l16;
            const float bn = bias[n];
            const int mbase = m0 + wm * 32 + mt * 16 + g * 8;
            #pragma unroll
            for (int r = 0; r < 8; ++r) {
                const int m = mbase + r;
                float v = c[mt][nt][r] + bn;
                if (EPI == 1) v = fmaxf(v, 0.f);
                if (EPI == 2) v += res[m * N + n];
                out[m * N + n] = v;
            }
        }
    }
}

// ---------------------------------------------------------------------------
// LayerNorm over 512 features; one WG of 256 threads per row.
// ---------------------------------------------------------------------------
__global__ __launch_bounds__(256)
void ln512(const float* __restrict__ x, const float* __restrict__ gamma,
           const float* __restrict__ beta, float* __restrict__ y)
{
    __shared__ float s1[256];
    __shared__ float s2[256];
    const int row = blockIdx.x, t = threadIdx.x;
    const float* xr = x + (size_t)row * DMODEL;
    const float a = xr[t], b = xr[t + 256];
    s1[t] = a + b;
    s2[t] = a * a + b * b;
    __syncthreads();
    for (int off = 128; off; off >>= 1) {
        if (t < off) { s1[t] += s1[t + off]; s2[t] += s2[t + off]; }
        __syncthreads();
    }
    const float mean = s1[0] * (1.f / DMODEL);
    const float var  = s2[0] * (1.f / DMODEL) - mean * mean;
    const float rs   = rsqrtf(var + 1e-5f);
    float* yr = y + (size_t)row * DMODEL;
    yr[t]       = (a - mean) * rs * gamma[t]       + beta[t];
    yr[t + 256] = (b - mean) * rs * gamma[t + 256] + beta[t + 256];
}

// ---------------------------------------------------------------------------
// Per-token gathered head (the HBM-bound stage): one WG per token streams
// HW1[se] (512 KB) with b128 loads. Layout: 64 column-quads (c) x 4 d-slices
// (s); lane (s,c) loads HW1[d, 4c..4c+3] for d = s, s+4, ... -> every global
// access is global_load_b128, 1KB per d-row across the WG, fully coalesced.
// ---------------------------------------------------------------------------
__global__ __launch_bounds__(256)
void head_kernel(const float* __restrict__ proj,
                 const float* __restrict__ HW1, const float* __restrict__ Hb1,
                 const float* __restrict__ HW2, const float* __restrict__ Hb2,
                 const int* __restrict__ input_ids, const int* __restrict__ head_id,
                 const int* __restrict__ num_sp, float* __restrict__ out)
{
    __shared__ float sp[DMODEL];
    __shared__ float part[4 * HHID];   // per-slice partial sums
    __shared__ float sh[HHID];
    __shared__ float slog[8];

    const int t   = blockIdx.x;
    const int tid = threadIdx.x;
    const int id  = input_ids[t];
    const int e   = head_id[id];
    const int se  = e > 0 ? e : 0;
    const int ns  = num_sp[id];

    const float* pr = proj + (size_t)t * DMODEL;
    sp[tid]       = pr[tid];
    sp[tid + 256] = pr[tid + 256];
    __syncthreads();

    const int c = tid & 63;            // column quad: cols 4c..4c+3
    const int s = tid >> 6;            // d-slice 0..3

    const float* wq = HW1 + (size_t)se * DMODEL * HHID + 4 * c;
    float ax = 0.f, ay = 0.f, az = 0.f, aw = 0.f;
    for (int d0 = s; d0 < DMODEL; d0 += 64) {
        // speculative prefetch ~96KB ahead of this lane's stream (OOB -> dropped)
        __builtin_prefetch(wq + (size_t)(d0 + 96) * HHID, 0, 0);
        #pragma unroll
        for (int i = 0; i < 16; ++i) {
            const int d = d0 + i * 4;
            const v4f w4 = *(const v4f*)(wq + (size_t)d * HHID);
            const float p = sp[d];
            ax = fmaf(p, w4.x, ax);
            ay = fmaf(p, w4.y, ay);
            az = fmaf(p, w4.z, az);
            aw = fmaf(p, w4.w, aw);
        }
    }
    part[s * HHID + 4 * c + 0] = ax;
    part[s * HHID + 4 * c + 1] = ay;
    part[s * HHID + 4 * c + 2] = az;
    part[s * HHID + 4 * c + 3] = aw;
    __syncthreads();

    // h[j] = relu(sum over slices + bias), j = tid
    {
        const int j = tid;
        float hv = part[j] + part[HHID + j] + part[2 * HHID + j] + part[3 * HHID + j];
        hv += Hb1[(size_t)se * HHID + j];
        sh[j] = fmaxf(hv, 0.f);
    }
    __syncthreads();

    // logits[k] via one wave32 per k (7 of 8 waves), shuffle reduction
    const int wv = tid >> 5, lane = tid & 31;
    if (wv < KSP) {
        const float* w2 = HW2 + (size_t)se * HHID * KSP + wv;   // stride KSP
        float a = 0.f;
        #pragma unroll
        for (int j = lane; j < HHID; j += 32)
            a = fmaf(sh[j], w2[(size_t)j * KSP], a);
        for (int off = 16; off; off >>= 1)
            a += __shfl_down(a, off, 32);
        if (lane == 0) slog[wv] = a + Hb2[(size_t)se * KSP + wv];
    }
    __syncthreads();

    // masked softmax + default/pad predicate (trivial serial over 7)
    if (tid == 0) {
        float l[KSP];
        float mx = -3.0e38f;
        #pragma unroll
        for (int k = 0; k < KSP; ++k) {
            const float v = (k < ns) ? slog[k] : -1e30f;
            l[k] = v;
            mx = fmaxf(mx, v);
        }
        float ssum = 0.f;
        #pragma unroll
        for (int k = 0; k < KSP; ++k) { l[k] = expf(l[k] - mx); ssum += l[k]; }
        const float inv = 1.f / ssum;
        const bool has = (e >= 0);
        const bool pad = (id == 0);
        float* o = out + (size_t)t * KSP;
        #pragma unroll
        for (int k = 0; k < KSP; ++k) {
            const float p = (k < ns) ? l[k] * inv : -1.f;
            const float dflt = (k == 0) ? 1.f : -1.f;
            float v = has ? p : dflt;
            if (pad) v = -1.f;
            o[k] = v;
        }
    }
}

// ---------------------------------------------------------------------------
extern "C" void kernel_launch(void* const* d_in, const int* in_sizes, int n_in,
                              void* d_out, int out_size, void* d_ws, size_t ws_size,
                              hipStream_t stream)
{
    const float* hidden = (const float*)d_in[0];
    const float* W_sh   = (const float*)d_in[1];
    const float* b_sh   = (const float*)d_in[2];
    const float* g_ln1  = (const float*)d_in[3];
    const float* be_ln1 = (const float*)d_in[4];
    const float* W_f1   = (const float*)d_in[5];
    const float* b_f1   = (const float*)d_in[6];
    const float* W_f2   = (const float*)d_in[7];
    const float* b_f2   = (const float*)d_in[8];
    const float* g_ln2  = (const float*)d_in[9];
    const float* be_ln2 = (const float*)d_in[10];
    const float* HW1    = (const float*)d_in[11];
    const float* Hb1    = (const float*)d_in[12];
    const float* HW2    = (const float*)d_in[13];
    const float* Hb2    = (const float*)d_in[14];
    const int* input_ids = (const int*)d_in[15];
    const int* head_id   = (const int*)d_in[16];
    const int* num_sp    = (const int*)d_in[17];
    float* out = (float*)d_out;

    // Workspace: 4 x [1024,512] fp32 buffers (8 MB total)
    float* t1     = (float*)d_ws;
    float* shrink = t1 + (size_t)TOKENS * DMODEL;
    float* ff1    = shrink + (size_t)TOKENS * DMODEL;
    float* t2     = ff1 + (size_t)TOKENS * DMODEL;
    float* proj   = t1;                         // t1 dead after LN1

    const dim3 blk(256);
    const dim3 ggrid(DMODEL / 64, TOKENS / 128);   // 8 x 8 WGs

    gemm_wmma<0><<<ggrid, blk, 0, stream>>>(hidden, W_sh, b_sh, nullptr, t1, 768);
    ln512<<<TOKENS, 256, 0, stream>>>(t1, g_ln1, be_ln1, shrink);
    gemm_wmma<1><<<ggrid, blk, 0, stream>>>(shrink, W_f1, b_f1, nullptr, ff1, DMODEL);
    gemm_wmma<2><<<ggrid, blk, 0, stream>>>(ff1, W_f2, b_f2, shrink, t2, DMODEL);
    ln512<<<TOKENS, 256, 0, stream>>>(t2, g_ln2, be_ln2, proj);

    head_kernel<<<TOKENS, 256, 0, stream>>>(proj, HW1, Hb1, HW2, Hb2,
                                            input_ids, head_id, num_sp, out);
}
